// Motif_Generation_dfs_35588099014824
// MI455X (gfx1250) — matrix operand order repaired and checked by reading, not compiled, not run on hardware
//
#include <hip/hip_runtime.h>
#include <hip/hip_bf16.h>

#define Bn 512
#define Tn 64
#define Hn 512
#define Vn 780
#define VP 832              // W_o columns padded to 13 groups of 64
#define N_AT 40
#define KC 4
#define MAX_NB 8
#define NMSG (Bn * Tn + 1)
#define BT (Bn * Tn)
#define RSTOP (BT + Bn)
#define NT_H (Hn / 16)      // 32 N-tiles for N=512
#define NT_VP (VP / 16)     // 52 N-tiles for the vocab head

typedef __attribute__((ext_vector_type(16))) __bf16 v16bf;
typedef __attribute__((ext_vector_type(8)))  __bf16 v8bf;
typedef __attribute__((ext_vector_type(8)))  float  v8f;

// ---------------- WMMA tile helpers (wave32, 16x16x32 bf16 -> f32) ----------
// A 16x32 (16-bit): lanes 0-15 row M=lane hold K 0..7 & 16..23; lanes 16-31 hold K 8..15 & 24..31
// Each lane's fragment is two contiguous, naturally aligned 8-element runs -> force b128 loads.
__device__ __forceinline__ v16bf ldA_bf(const __bf16* rowbase, int kb) {
  v8bf lo = *(const v8bf*)(rowbase + kb);
  v8bf hi = *(const v8bf*)(rowbase + kb + 16);
  return __builtin_shufflevector(lo, hi, 0, 1, 2, 3, 4, 5, 6, 7, 8, 9, 10, 11, 12, 13, 14, 15);
}
__device__ __forceinline__ v16bf ldA_f32(const float* rowbase, int kb) {
  v8f lo = *(const v8f*)(rowbase + kb);
  v8f hi = *(const v8f*)(rowbase + kb + 16);
  v16bf a;
#pragma unroll
  for (int j = 0; j < 8; ++j) { a[j] = (__bf16)lo[j]; a[8 + j] = (__bf16)hi[j]; }
  return a;
}
// B fragments pre-swizzled: fragment (kc,ntile) holds 32 lanes x 16 contiguous bf16.
// One aligned 32-byte vector load per lane.
__device__ __forceinline__ v16bf ldBfrag(const __bf16* base, int frag, int lane) {
  return *(const v16bf*)(base + ((size_t)frag << 9) + (lane << 4));
}
__device__ __forceinline__ v8f wmma_bf(v16bf a, v16bf b, v8f c) {
  return __builtin_amdgcn_wmma_f32_16x16x32_bf16(false, a, false, b, (short)0, c, false, false);
}
__device__ __forceinline__ float sigf(float x) { return 1.f / (1.f + __expf(-x)); }

// ---------------- Prep / utility kernels -----------------------------------
__global__ void k_init(float* tableF, float* accum) {
  size_t i = (size_t)blockIdx.x * 256 + threadIdx.x;
  if (i < (size_t)NMSG * Hn) tableF[i] = 0.f;
  if (i < 8) accum[i] = 0.f;
}

// Swizzle row-major f32 weight [K2 x Nsrc] into bf16 WMMA B fragments [K2/32][N/16][32][16]
__global__ void k_swz(const float* __restrict__ src, __bf16* __restrict__ dst,
                      int K2, int N, int Nsrc) {
  int i = blockIdx.x * 256 + threadIdx.x;
  if (i >= K2 * N) return;
  int j = i & 15;
  int lane = (i >> 4) & 31;
  int fi = i >> 9;
  int ntiles = N >> 4;
  int kc = fi / ntiles, ntile = fi % ntiles;
  int n = (ntile << 4) | (lane & 15);
  int k = (kc << 5) + ((lane >> 4) << 3) + (j & 7) + ((j >> 3) << 4);
  dst[i] = (n < Nsrc) ? (__bf16)src[(size_t)k * Nsrc + n] : (__bf16)0.f;
}

// x_all[b*T+t, h] = sum_j node_rep[b, clique_idx[b,t,j], h]  (bf16)
__global__ void k_xall(const float* __restrict__ node_rep, const int* __restrict__ clique,
                       __bf16* __restrict__ xall) {
  size_t i = (size_t)blockIdx.x * 256 + threadIdx.x;
  if (i >= (size_t)BT * Hn) return;
  int h = (int)(i & (Hn - 1));
  int bt = (int)(i >> 9);
  int b = bt >> 6;
  float s = 0.f;
#pragma unroll
  for (int j = 0; j < KC; ++j) {
    int a = clique[bt * KC + j];
    s += node_rep[((size_t)b * N_AT + a) * Hn + h];
  }
  xall[i] = (__bf16)s;
}

__global__ void k_rootx(const float* __restrict__ node_rep, const int* __restrict__ rci,
                        __bf16* __restrict__ rootx) {
  int i = blockIdx.x * 256 + threadIdx.x;
  if (i >= Bn * Hn) return;
  int b = i >> 9, h = i & (Hn - 1);
  float s = 0.f;
#pragma unroll
  for (int j = 0; j < KC; ++j) s += node_rep[((size_t)b * N_AT + rci[b * KC + j]) * Hn + h];
  rootx[i] = (__bf16)s;
}

// Per-step: sum_h (f32) and o (bf16, scattered into o_all) gather-sums from table
__global__ void k_stepS(const float* __restrict__ tableF, const int* __restrict__ nei_h,
                        const int* __restrict__ nei_o, float* __restrict__ shF,
                        __bf16* __restrict__ oall, int t) {
  int i = blockIdx.x * 256 + threadIdx.x;
  if (i >= Bn * Hn) return;
  int b = i >> 9, h = i & (Hn - 1);
  float s = 0.f, o = 0.f;
#pragma unroll
  for (int nb = 0; nb < MAX_NB; ++nb) {
    s += tableF[(size_t)nei_h[((size_t)b * Tn + t) * MAX_NB + nb] * Hn + h];
    o += tableF[(size_t)nei_o[((size_t)b * Tn + t) * MAX_NB + nb] * Hn + h];
  }
  shF[i] = s;
  oall[((size_t)b * Tn + t) * Hn + h] = (__bf16)o;
}

// ---------------- GRU step kernel A: z, r, sum_g (16 rows x 64 cols / wave) -
__global__ __launch_bounds__(32) void k_stepA(
    const __bf16* __restrict__ xall, const float* __restrict__ shF,
    const float* __restrict__ tableF, const int* __restrict__ nei_h,
    const __bf16* __restrict__ WzB, const float* __restrict__ Wzb,
    const __bf16* __restrict__ WrB, const float* __restrict__ Wrb,
    const __bf16* __restrict__ UrB, float* __restrict__ zF, float* __restrict__ sgF, int t) {
  const int lane = threadIdx.x;
  const int b0 = (blockIdx.x >> 3) << 4;       // 32 row tiles
  const int g = blockIdx.x & 7;                // 8 col groups of 64
  const int n0 = g << 6;
  const int nt0 = g << 2;
  const int m = lane & 15;
  const int kb = (lane >> 4) << 3;
  const int n = lane & 15;
  const int mbase = (lane >> 4) << 3;

  const __bf16* xrow = xall + ((size_t)(b0 + m) * Tn + t) * Hn;
  const float* srow = shF + (size_t)(b0 + m) * Hn;

  // z = sigmoid([x | sum_h] @ Wz + b)
  v8f cz[4] = {};
  for (int kc = 0; kc < 16; ++kc) {
    v16bf a = ldA_bf(xrow + (kc << 5), kb);
#pragma unroll
    for (int nt = 0; nt < 4; ++nt)
      cz[nt] = wmma_bf(a, ldBfrag(WzB, kc * NT_H + nt0 + nt, lane), cz[nt]);
  }
  for (int kc = 0; kc < 16; ++kc) {
    v16bf a = ldA_f32(srow + (kc << 5), kb);
#pragma unroll
    for (int nt = 0; nt < 4; ++nt)
      cz[nt] = wmma_bf(a, ldBfrag(WzB, (16 + kc) * NT_H + nt0 + nt, lane), cz[nt]);
  }

  // x @ Wr
  v8f cx[4] = {};
  for (int kc = 0; kc < 16; ++kc) {
    v16bf a = ldA_bf(xrow + (kc << 5), kb);
#pragma unroll
    for (int nt = 0; nt < 4; ++nt)
      cx[nt] = wmma_bf(a, ldBfrag(WrB, kc * NT_H + nt0 + nt, lane), cx[nt]);
  }

  float bz[4], br[4];
#pragma unroll
  for (int nt = 0; nt < 4; ++nt) {
    bz[nt] = Wzb[n0 + (nt << 4) + n];
    br[nt] = Wrb[n0 + (nt << 4) + n];
  }
  float sg[4][8];
#pragma unroll
  for (int nt = 0; nt < 4; ++nt)
#pragma unroll
    for (int r = 0; r < 8; ++r) sg[nt][r] = 0.f;

  for (int nb = 0; nb < MAX_NB; ++nb) {
    const int idxA = nei_h[((size_t)(b0 + m) * Tn + t) * MAX_NB + nb];
    const float* hrow = tableF + (size_t)idxA * Hn;
    v8f cr[4] = {};
    for (int kc = 0; kc < 16; ++kc) {
      v16bf a = ldA_f32(hrow + (kc << 5), kb);
#pragma unroll
      for (int nt = 0; nt < 4; ++nt)
        cr[nt] = wmma_bf(a, ldBfrag(UrB, kc * NT_H + nt0 + nt, lane), cr[nt]);
    }
#pragma unroll
    for (int r = 0; r < 8; ++r) {
      const int row = b0 + mbase + r;
      const int idxD = nei_h[((size_t)row * Tn + t) * MAX_NB + nb];
      const float* drow = tableF + (size_t)idxD * Hn;
#pragma unroll
      for (int nt = 0; nt < 4; ++nt) {
        const float hv = drow[n0 + (nt << 4) + n];
        const float rv = sigf(cx[nt][r] + br[nt] + cr[nt][r]);
        sg[nt][r] += rv * hv;
      }
    }
  }
#pragma unroll
  for (int nt = 0; nt < 4; ++nt)
#pragma unroll
    for (int r = 0; r < 8; ++r) {
      const int row = b0 + mbase + r;
      const int col = n0 + (nt << 4) + n;
      zF[(size_t)row * Hn + col] = sigf(cz[nt][r] + bz[nt]);
      sgF[(size_t)row * Hn + col] = sg[nt][r];
    }
}

// ---------------- GRU step kernel B: pre_h, new_h, scatter ------------------
__global__ __launch_bounds__(32) void k_stepB(
    const __bf16* __restrict__ xall, const float* __restrict__ sgF,
    const float* __restrict__ zF, const float* __restrict__ shF,
    const __bf16* __restrict__ WhB, const float* __restrict__ Whb,
    float* __restrict__ tableF, int t) {
  const int lane = threadIdx.x;
  const int b0 = (blockIdx.x >> 3) << 4;
  const int g = blockIdx.x & 7;
  const int n0 = g << 6;
  const int nt0 = g << 2;
  const int m = lane & 15;
  const int kb = (lane >> 4) << 3;
  const int n = lane & 15;
  const int mbase = (lane >> 4) << 3;

  const __bf16* xrow = xall + ((size_t)(b0 + m) * Tn + t) * Hn;
  const float* grow = sgF + (size_t)(b0 + m) * Hn;

  v8f c[4] = {};
  for (int kc = 0; kc < 16; ++kc) {
    v16bf a = ldA_bf(xrow + (kc << 5), kb);
#pragma unroll
    for (int nt = 0; nt < 4; ++nt)
      c[nt] = wmma_bf(a, ldBfrag(WhB, kc * NT_H + nt0 + nt, lane), c[nt]);
  }
  for (int kc = 0; kc < 16; ++kc) {
    v16bf a = ldA_f32(grow + (kc << 5), kb);
#pragma unroll
    for (int nt = 0; nt < 4; ++nt)
      c[nt] = wmma_bf(a, ldBfrag(WhB, (16 + kc) * NT_H + nt0 + nt, lane), c[nt]);
  }

#pragma unroll
  for (int nt = 0; nt < 4; ++nt) {
    const float bias = Whb[n0 + (nt << 4) + n];
#pragma unroll
    for (int r = 0; r < 8; ++r) {
      const int row = b0 + mbase + r;
      const int col = n0 + (nt << 4) + n;
      const float ph = tanhf(c[nt][r] + bias);
      const float zv = zF[(size_t)row * Hn + col];
      const float shv = shF[(size_t)row * Hn + col];
      tableF[((size_t)row * Tn + t + 1) * Hn + col] = (1.f - zv) * shv + zv * ph;
    }
  }
}

// ---------------- Root o gather (after scan) --------------------------------
__global__ void k_rooto(const float* __restrict__ tableF, const int* __restrict__ root_nei,
                        __bf16* __restrict__ rooto) {
  int i = blockIdx.x * 256 + threadIdx.x;
  if (i >= Bn * Hn) return;
  int b = i >> 9, h = i & (Hn - 1);
  float o = 0.f;
#pragma unroll
  for (int nb = 0; nb < MAX_NB; ++nb)
    o += tableF[(size_t)root_nei[b * MAX_NB + nb] * Hn + h];
  rooto[i] = (__bf16)o;
}

// ---------------- Stop head: relu([x|o] @ U + b) @ Us + loss ---------------
__global__ __launch_bounds__(32) void k_stop(
    const __bf16* __restrict__ xall, const __bf16* __restrict__ oall,
    const __bf16* __restrict__ rootx, const __bf16* __restrict__ rooto,
    const __bf16* __restrict__ UB_, const float* __restrict__ Ub,
    const float* __restrict__ Usw, const float* __restrict__ Usb,
    const int* __restrict__ direction, float* __restrict__ accum) {
  const int lane = threadIdx.x;
  const int row0 = blockIdx.x << 4;
  const int m = lane & 15;
  const int kb = (lane >> 4) << 3;
  const int n = lane & 15;
  const int mbase = (lane >> 4) << 3;

  const int rowm = row0 + m;
  const __bf16* arow1 = (rowm < BT) ? (xall + (size_t)rowm * Hn) : (rootx + (size_t)(rowm - BT) * Hn);
  const __bf16* arow2 = (rowm < BT) ? (oall + (size_t)rowm * Hn) : (rooto + (size_t)(rowm - BT) * Hn);

  float racc[8];
#pragma unroll
  for (int r = 0; r < 8; ++r) racc[r] = 0.f;

  for (int g = 0; g < 8; ++g) {
    v8f c[4] = {};
    for (int kc = 0; kc < 16; ++kc) {
      v16bf a = ldA_bf(arow1 + (kc << 5), kb);
#pragma unroll
      for (int nt = 0; nt < 4; ++nt)
        c[nt] = wmma_bf(a, ldBfrag(UB_, kc * NT_H + (g << 2) + nt, lane), c[nt]);
    }
    for (int kc = 0; kc < 16; ++kc) {
      v16bf a = ldA_bf(arow2 + (kc << 5), kb);
#pragma unroll
      for (int nt = 0; nt < 4; ++nt)
        c[nt] = wmma_bf(a, ldBfrag(UB_, (16 + kc) * NT_H + (g << 2) + nt, lane), c[nt]);
    }
#pragma unroll
    for (int nt = 0; nt < 4; ++nt) {
      const int col = (g << 6) + (nt << 4) + n;
      const float bias = Ub[col];
      const float us = Usw[col];
#pragma unroll
      for (int r = 0; r < 8; ++r) {
        float hv = c[nt][r] + bias;
        hv = hv > 0.f ? hv : 0.f;
        racc[r] += hv * us;
      }
    }
  }
#pragma unroll
  for (int r = 0; r < 8; ++r) {
    float v = racc[r];
#pragma unroll
    for (int mask = 1; mask < 16; mask <<= 1) v += __shfl_xor(v, mask, 32);
    racc[r] = v;
  }
  if (n == 0) {
    const float usb = Usb[0];
    float lloss = 0.f, lcorr = 0.f;
#pragma unroll
    for (int r = 0; r < 8; ++r) {
      const int row = row0 + mbase + r;
      const float s = racc[r] + usb;
      const float tgt = (row < BT) ? (float)direction[row] : 0.f;
      lloss += fmaxf(s, 0.f) + log1pf(__expf(-fabsf(s))) - s * tgt;
      const float p = (s >= 0.f) ? 1.f : 0.f;
      lcorr += (p == tgt) ? 1.f : 0.f;
    }
    atomicAdd(&accum[1], lloss);
    atomicAdd(&accum[4], lcorr);
  }
}

// ---------------- Pred head pass 1: hmid = relu(new_h @ W + b) --------------
__global__ __launch_bounds__(32) void k_mid(const float* __restrict__ tableF,
                                            const __bf16* __restrict__ WB_,
                                            const float* __restrict__ Wb,
                                            __bf16* __restrict__ hmid) {
  const int lane = threadIdx.x;
  const int row0 = (blockIdx.x >> 3) << 4;
  const int g = blockIdx.x & 7;
  const int n0 = g << 6;
  const int m = lane & 15;
  const int kb = (lane >> 4) << 3;
  const int n = lane & 15;
  const int mbase = (lane >> 4) << 3;

  const float* arow = tableF + (size_t)(row0 + m + 1) * Hn;  // new_h row i == table row i+1
  v8f c[4] = {};
  for (int kc = 0; kc < 16; ++kc) {
    v16bf a = ldA_f32(arow + (kc << 5), kb);
#pragma unroll
    for (int nt = 0; nt < 4; ++nt)
      c[nt] = wmma_bf(a, ldBfrag(WB_, kc * NT_H + (g << 2) + nt, lane), c[nt]);
  }
#pragma unroll
  for (int nt = 0; nt < 4; ++nt) {
    const float bias = Wb[n0 + (nt << 4) + n];
#pragma unroll
    for (int r = 0; r < 8; ++r) {
      float hv = c[nt][r] + bias;
      hv = hv > 0.f ? hv : 0.f;
      hmid[(size_t)(row0 + mbase + r) * Hn + n0 + (nt << 4) + n] = (__bf16)hv;
    }
  }
}

// ---------------- Pred head pass 2: logits GEMM + online softmax/CE ---------
__global__ __launch_bounds__(32) void k_pred(const __bf16* __restrict__ hmid,
                                             const __bf16* __restrict__ WoB,
                                             const float* __restrict__ Wob,
                                             const int* __restrict__ pred_target,
                                             const int* __restrict__ direction,
                                             float* __restrict__ accum) {
  const int lane = threadIdx.x;
  const int row0 = blockIdx.x << 4;
  const int m = lane & 15;
  const int kb = (lane >> 4) << 3;
  const int n = lane & 15;
  const int mbase = (lane >> 4) << 3;

  const __bf16* arow = hmid + (size_t)(row0 + m) * Hn;
  int tg[8];
  float rmax[8], rsum[8], rbestv[8], rtgt[8];
  int rbest[8];
#pragma unroll
  for (int r = 0; r < 8; ++r) {
    tg[r] = pred_target[row0 + mbase + r];
    rmax[r] = -1e30f; rsum[r] = 0.f; rbestv[r] = -1e30f; rbest[r] = 0; rtgt[r] = 0.f;
  }

  for (int g = 0; g < 13; ++g) {  // 13 groups of 64 cols (VP = 832)
    v8f c[4] = {};
    for (int kc = 0; kc < 16; ++kc) {
      v16bf a = ldA_bf(arow + (kc << 5), kb);
#pragma unroll
      for (int nt = 0; nt < 4; ++nt)
        c[nt] = wmma_bf(a, ldBfrag(WoB, kc * NT_VP + (g << 2) + nt, lane), c[nt]);
    }
#pragma unroll
    for (int nt = 0; nt < 4; ++nt) {
      const int n0 = (g << 6) + (nt << 4);
      const int col = n0 + n;
      const bool valid = col < Vn;
      const float bias = valid ? Wob[col] : 0.f;
#pragma unroll
      for (int r = 0; r < 8; ++r) {
        const float l = valid ? (c[nt][r] + bias) : -1e30f;
        // tile row max across the 16 lanes of this half
        float tm = l;
#pragma unroll
        for (int mask = 1; mask < 16; mask <<= 1) tm = fmaxf(tm, __shfl_xor(tm, mask, 32));
        const float nm = fmaxf(rmax[r], tm);
        float e = __expf(l - nm);
#pragma unroll
        for (int mask = 1; mask < 16; mask <<= 1) e += __shfl_xor(e, mask, 32);
        rsum[r] = rsum[r] * __expf(rmax[r] - nm) + e;
        rmax[r] = nm;
        // tile argmax (first-max tie-break)
        float bv = l;
        int bi = col;
#pragma unroll
        for (int mask = 1; mask < 16; mask <<= 1) {
          const float ov = __shfl_xor(bv, mask, 32);
          const int oi = __shfl_xor(bi, mask, 32);
          if (ov > bv || (ov == bv && oi < bi)) { bv = ov; bi = oi; }
        }
        if (bv > rbestv[r] || (bv == rbestv[r] && bi < rbest[r])) { rbestv[r] = bv; rbest[r] = bi; }
        // target logit (non-divergent shuffle, conditional keep)
        const int d = tg[r] - n0;
        const int src = (lane & 16) | (d & 15);
        const float cand = __shfl(l, src, 32);
        if (d >= 0 && d < 16) rtgt[r] = cand;
      }
    }
  }
  if (n == 0) {
    float lloss = 0.f, lcorr = 0.f, lpm = 0.f;
#pragma unroll
    for (int r = 0; r < 8; ++r) {
      const int row = row0 + mbase + r;
      const float pm = (direction[row] == 1) ? 1.f : 0.f;
      const float ce = rmax[r] + __logf(rsum[r]) - rtgt[r];
      lloss += ce * pm;
      lcorr += (rbest[r] == tg[r]) ? pm : 0.f;
      lpm += pm;
    }
    atomicAdd(&accum[0], lloss);
    atomicAdd(&accum[2], lcorr);
    atomicAdd(&accum[3], lpm);
  }
}

__global__ void k_final(const float* __restrict__ accum, float* __restrict__ out) {
  if (blockIdx.x == 0 && threadIdx.x == 0) {
    out[0] = accum[0] / (float)Bn;        // pred_loss
    out[1] = accum[1] / (float)Bn;        // stop_loss
    out[2] = accum[2] / accum[3];         // pred_acc
    out[3] = accum[4] / (float)RSTOP;     // stop_acc
  }
}

// ---------------------------------------------------------------------------
extern "C" void kernel_launch(void* const* d_in, const int* in_sizes, int n_in,
                              void* d_out, int out_size, void* d_ws, size_t ws_size,
                              hipStream_t stream) {
  const float* node_rep   = (const float*)d_in[0];
  const int*   clique_idx = (const int*)d_in[1];
  const int*   root_cliq  = (const int*)d_in[2];
  const int*   nei_h      = (const int*)d_in[3];
  const int*   nei_o      = (const int*)d_in[4];
  const int*   root_nei   = (const int*)d_in[5];
  const int*   direction  = (const int*)d_in[6];
  const int*   pred_tgt   = (const int*)d_in[7];
  const float* Wzw = (const float*)d_in[8];
  const float* Wzb = (const float*)d_in[9];
  const float* Wrw = (const float*)d_in[10];
  const float* Wrb = (const float*)d_in[11];
  const float* Urw = (const float*)d_in[12];
  const float* Whw = (const float*)d_in[13];
  const float* Whb = (const float*)d_in[14];
  const float* Ww  = (const float*)d_in[15];
  const float* Wb  = (const float*)d_in[16];
  const float* Uw  = (const float*)d_in[17];
  const float* Ubi = (const float*)d_in[18];
  const float* Wow = (const float*)d_in[19];
  const float* Wob = (const float*)d_in[20];
  const float* Usw = (const float*)d_in[21];
  const float* Usb = (const float*)d_in[22];

  char* w = (char*)d_ws;
  size_t off = 0;
  auto alloc = [&](size_t bytes) -> void* {
    void* p = w + off;
    off = (off + bytes + 255) & ~(size_t)255;
    return p;
  };
  float*  tableF = (float*)alloc((size_t)NMSG * Hn * sizeof(float));
  __bf16* xallB  = (__bf16*)alloc((size_t)BT * Hn * sizeof(__bf16));
  __bf16* oallB  = (__bf16*)alloc((size_t)BT * Hn * sizeof(__bf16));
  __bf16* hmidB  = (__bf16*)alloc((size_t)BT * Hn * sizeof(__bf16));
  __bf16* rootxB = (__bf16*)alloc((size_t)Bn * Hn * sizeof(__bf16));
  __bf16* rootoB = (__bf16*)alloc((size_t)Bn * Hn * sizeof(__bf16));
  float*  shF    = (float*)alloc((size_t)Bn * Hn * sizeof(float));
  float*  zF     = (float*)alloc((size_t)Bn * Hn * sizeof(float));
  float*  sgF    = (float*)alloc((size_t)Bn * Hn * sizeof(float));
  __bf16* WzB = (__bf16*)alloc((size_t)2 * Hn * Hn * sizeof(__bf16));
  __bf16* WrB = (__bf16*)alloc((size_t)Hn * Hn * sizeof(__bf16));
  __bf16* UrB = (__bf16*)alloc((size_t)Hn * Hn * sizeof(__bf16));
  __bf16* WhB = (__bf16*)alloc((size_t)2 * Hn * Hn * sizeof(__bf16));
  __bf16* WB_ = (__bf16*)alloc((size_t)Hn * Hn * sizeof(__bf16));
  __bf16* UB_ = (__bf16*)alloc((size_t)2 * Hn * Hn * sizeof(__bf16));
  __bf16* WoB = (__bf16*)alloc((size_t)Hn * VP * sizeof(__bf16));
  float*  accum = (float*)alloc(8 * sizeof(float));

  // init table + accumulators
  {
    size_t nel = (size_t)NMSG * Hn;
    k_init<<<(unsigned)((nel + 255) / 256), 256, 0, stream>>>(tableF, accum);
  }
  // features
  k_xall<<<(unsigned)(((size_t)BT * Hn + 255) / 256), 256, 0, stream>>>(node_rep, clique_idx, xallB);
  k_rootx<<<(Bn * Hn + 255) / 256, 256, 0, stream>>>(node_rep, root_cliq, rootxB);
  // weight swizzles into WMMA B-fragment layout
  k_swz<<<(2 * Hn * Hn + 255) / 256, 256, 0, stream>>>(Wzw, WzB, 2 * Hn, Hn, Hn);
  k_swz<<<(Hn * Hn + 255) / 256, 256, 0, stream>>>(Wrw, WrB, Hn, Hn, Hn);
  k_swz<<<(Hn * Hn + 255) / 256, 256, 0, stream>>>(Urw, UrB, Hn, Hn, Hn);
  k_swz<<<(2 * Hn * Hn + 255) / 256, 256, 0, stream>>>(Whw, WhB, 2 * Hn, Hn, Hn);
  k_swz<<<(Hn * Hn + 255) / 256, 256, 0, stream>>>(Ww, WB_, Hn, Hn, Hn);
  k_swz<<<(2 * Hn * Hn + 255) / 256, 256, 0, stream>>>(Uw, UB_, 2 * Hn, Hn, Hn);
  k_swz<<<(Hn * VP + 255) / 256, 256, 0, stream>>>(Wow, WoB, Hn, VP, Vn);

  // sequential tree-GRU scan
  const int tileBlocks = (Bn / 16) * (Hn / 64);  // 32 * 8 = 256
  for (int t = 0; t < Tn; ++t) {
    k_stepS<<<(Bn * Hn + 255) / 256, 256, 0, stream>>>(tableF, nei_h, nei_o, shF, oallB, t);
    k_stepA<<<tileBlocks, 32, 0, stream>>>(xallB, shF, tableF, nei_h, WzB, Wzb, WrB, Wrb,
                                           UrB, zF, sgF, t);
    k_stepB<<<tileBlocks, 32, 0, stream>>>(xallB, sgF, zF, shF, WhB, Whb, tableF, t);
  }

  // heads
  k_rooto<<<(Bn * Hn + 255) / 256, 256, 0, stream>>>(tableF, root_nei, rootoB);
  k_stop<<<RSTOP / 16, 32, 0, stream>>>(xallB, oallB, rootxB, rootoB, UB_, Ubi, Usw, Usb,
                                        direction, accum);
  k_mid<<<(BT / 16) * (Hn / 64), 32, 0, stream>>>(tableF, WB_, Wb, hmidB);
  k_pred<<<BT / 16, 32, 0, stream>>>(hmidB, WoB, Wob, pred_tgt, direction, accum);
  k_final<<<1, 1, 0, stream>>>(accum, (float*)d_out);
}
